// GCNModelVAE_17463337026205
// MI455X (gfx1250) — matrix operand order repaired and checked by reading, not compiled, not run on hardware
//
#include <hip/hip_runtime.h>
#include <hip/hip_bf16.h>

// ---------------------------------------------------------------------------
// GCN-VAE forward on gfx1250 via v_wmma_f32_16x16x32_bf16
//   t0T = (x@W1)^T          -> bf16 [H1,N]   (transposed vector-store epilogue)
//   hT  = relu(adj@t0)^T    -> bf16 [H1,N]   (TRANSB fast staging, both ops b128)
//   hmT = WmT @ hT          -> bf16 [H2,N]
//   hsT = WsT @ hT          -> bf16 [H2,N]
//   zm  = adj @ hm          -> f32  [N,H2]   (TRANSB fast staging)
//   zs  = adj @ hs          -> f32  [N,H2]
//   z   = zm + eps*exp(zs)  -> bf16 [N,H2]
//   out = sigmoid(z @ z^T)  -> f32  [N,N]    (symmetric => transposed b128 stores)
// ---------------------------------------------------------------------------

typedef __attribute__((ext_vector_type(16))) __bf16 v16bf;
typedef __attribute__((ext_vector_type(8)))  float  v8f;

#define NN   8192
#define FIN  512
#define HH1  256
#define HH2  128

constexpr int BM = 128;
constexpr int BN = 128;
constexpr int BK = 32;
constexpr int LDSK = BK + 8;   // pad rows to 40 elems (80 B) vs bank conflicts

// ---------------------------------------------------------------------------
// fp32 -> bf16 conversion, 8 elems/thread (b128 in, packed cvt, b128 out)
// ---------------------------------------------------------------------------
__global__ void cvt_f32_bf16_kernel(const float* __restrict__ src,
                                    __bf16* __restrict__ dst, long n8) {
  long i = (long)blockIdx.x * blockDim.x + threadIdx.x;
  long stride = (long)gridDim.x * blockDim.x;
  for (; i < n8; i += stride) {
    const float* s = src + i * 8;
    float4 a = *(const float4*)(s);
    float4 b = *(const float4*)(s + 4);
    __bf16 o[8] __attribute__((aligned(16)));
    o[0] = (__bf16)a.x; o[1] = (__bf16)a.y; o[2] = (__bf16)a.z; o[3] = (__bf16)a.w;
    o[4] = (__bf16)b.x; o[5] = (__bf16)b.y; o[6] = (__bf16)b.z; o[7] = (__bf16)b.w;
    *(uint4*)(dst + i * 8) = *(const uint4*)o;
  }
}

// fp32 [R,C] -> bf16 transposed [C,R] (small weight matrices only)
__global__ void cvt_f32_bf16_T_kernel(const float* __restrict__ src,
                                      __bf16* __restrict__ dst, int R, int C) {
  int i = blockIdx.x * blockDim.x + threadIdx.x;
  int n = R * C;
  int stride = gridDim.x * blockDim.x;
  for (; i < n; i += stride) {
    int r = i / C, c = i % C;
    dst[(long)c * R + r] = (__bf16)src[i];
  }
}

// ---------------------------------------------------------------------------
// z = z_mean + eps * exp(log_std)   (fp32 math, bf16 out for the final GEMM)
// ---------------------------------------------------------------------------
__global__ void reparam_kernel(const float* __restrict__ zm,
                               const float* __restrict__ zs,
                               const float* __restrict__ eps,
                               __bf16* __restrict__ zb, long n) {
  long i = (long)blockIdx.x * blockDim.x + threadIdx.x;
  long stride = (long)gridDim.x * blockDim.x;
  for (; i < n; i += stride) {
    float z = zm[i] + eps[i] * __expf(zs[i]);
    zb[i] = (__bf16)z;
  }
}

// ---------------------------------------------------------------------------
// Tiled bf16 WMMA GEMM: C = epilogue(A @ B)
//   A: M x K row-major (bf16); B: K x N row-major, or TRANSB: Bsrc[N x K]
//   EPI: 0 = bf16           2 = f32
//        4 = bf16 transposed (C^T, contiguous b128/fragment)
//        5 = relu -> bf16 transposed
//        6 = sigmoid -> f32 transposed (valid when C is symmetric)
// Block = 256 threads (8 waves); block tile 128x128, wave tile 32x64.
// Ping-pong LDS + register prefetch: one barrier per k-step. All dims exact
// tile multiples, so EXEC stays all-ones around every WMMA.
// ---------------------------------------------------------------------------
template <int EPI, bool TRANSB>
__global__ __launch_bounds__(256) void gemm_bf16_wmma(
    const __bf16* __restrict__ A, const __bf16* __restrict__ B,
    void* __restrict__ Cout, int K, int lda, int ldb, int ldc) {
  __shared__ __attribute__((aligned(16))) __bf16 sA[2][BM][LDSK];   // [buf][m][k]
  __shared__ __attribute__((aligned(16))) __bf16 sB[2][BN][LDSK];   // [buf][n][k]

  const int tid  = threadIdx.x;
  const int lane = tid & 31;
  const int wid  = tid >> 5;
  const int wm   = wid & 3;    // wave row 0..3  -> 32 rows each
  const int wn   = wid >> 2;   // wave col 0..1  -> 64 cols each
  const int m0   = blockIdx.x * BM;
  const int n0   = blockIdx.y * BN;

  // per-thread staging coordinates (chunk c adds +64 rows / +16 k-rows)
  const int arow = tid >> 2;             // 0..63
  const int akc  = (tid & 3) << 3;       // 0,8,16,24
  const int bkr  = tid >> 4;             // 0..15
  const int bnc  = (tid & 15) << 3;      // 0..120

  v8f acc[2][4];
#pragma unroll
  for (int i = 0; i < 2; ++i)
#pragma unroll
    for (int j = 0; j < 4; ++j)
#pragma unroll
      for (int e = 0; e < 8; ++e) acc[i][j][e] = 0.0f;

  uint4 ra0, ra1, rb0, rb1;

  // ---- prefetch k0 = 0 ----
  ra0 = *(const uint4*)(A + (long)(m0 + arow) * lda + akc);
  ra1 = *(const uint4*)(A + (long)(m0 + arow + 64) * lda + akc);
  if (TRANSB) {
    rb0 = *(const uint4*)(B + (long)(n0 + arow) * ldb + akc);
    rb1 = *(const uint4*)(B + (long)(n0 + arow + 64) * ldb + akc);
  } else {
    rb0 = *(const uint4*)(B + (long)bkr * ldb + n0 + bnc);
    rb1 = *(const uint4*)(B + (long)(bkr + 16) * ldb + n0 + bnc);
  }
  // ---- stage into buffer 0 ----
  *(uint4*)(&sA[0][arow][akc])      = ra0;
  *(uint4*)(&sA[0][arow + 64][akc]) = ra1;
  if (TRANSB) {
    *(uint4*)(&sB[0][arow][akc])      = rb0;
    *(uint4*)(&sB[0][arow + 64][akc]) = rb1;
  } else {
    __bf16 t[8] __attribute__((aligned(16)));
    *(uint4*)t = rb0;
#pragma unroll
    for (int j = 0; j < 8; ++j) sB[0][bnc + j][bkr] = t[j];
    *(uint4*)t = rb1;
#pragma unroll
    for (int j = 0; j < 8; ++j) sB[0][bnc + j][bkr + 16] = t[j];
  }

  const int nsteps = K / BK;
  for (int s = 0; s < nsteps; ++s) {
    const int cur = s & 1;
    __syncthreads();   // buf[cur] fully staged; prior reads of buf[cur^1] done

    // ---- prefetch next k-chunk while this step computes ----
    const bool more = (s + 1) < nsteps;
    if (more) {
      const int k0 = (s + 1) * BK;
      ra0 = *(const uint4*)(A + (long)(m0 + arow) * lda + k0 + akc);
      ra1 = *(const uint4*)(A + (long)(m0 + arow + 64) * lda + k0 + akc);
      if (TRANSB) {
        rb0 = *(const uint4*)(B + (long)(n0 + arow) * ldb + k0 + akc);
        rb1 = *(const uint4*)(B + (long)(n0 + arow + 64) * ldb + k0 + akc);
      } else {
        rb0 = *(const uint4*)(B + (long)(k0 + bkr) * ldb + n0 + bnc);
        rb1 = *(const uint4*)(B + (long)(k0 + bkr + 16) * ldb + n0 + bnc);
      }
    }

    // ---- fragments (documented wave32 VGPR layouts) ----
    // A 16x32 bf16: lanes 0-15 rows M with K {0-7,16-23}; lanes 16-31 {8-15,24-31}
    const int r  = lane & 15;
    const int kh = (lane >> 4) << 3;      // 0 or 8
    v16bf af[2];
#pragma unroll
    for (int i = 0; i < 2; ++i) {
      const __bf16* p = &sA[cur][wm * 32 + i * 16 + r][0];
#pragma unroll
      for (int j = 0; j < 8; ++j) {
        af[i][j]     = p[kh + j];
        af[i][8 + j] = p[16 + kh + j];
      }
    }
    // B 32x16 bf16: lanes 0-15 cols N with K {0-7,16-23}; lanes 16-31 {8-15,24-31}
    const int kb = (lane >> 4) << 3;      // 0 or 8
    v16bf bfv[4];
#pragma unroll
    for (int j4 = 0; j4 < 4; ++j4) {
      const __bf16* p = &sB[cur][wn * 64 + j4 * 16 + r][0];
#pragma unroll
      for (int j = 0; j < 8; ++j) {
        bfv[j4][j]     = p[kb + j];
        bfv[j4][8 + j] = p[16 + kb + j];
      }
    }

#pragma unroll
    for (int i = 0; i < 2; ++i)
#pragma unroll
      for (int j = 0; j < 4; ++j)
        acc[i][j] = __builtin_amdgcn_wmma_f32_16x16x32_bf16(
            false, af[i], false, bfv[j], (short)0, acc[i][j], false, false);

    // ---- stage prefetched chunk into the other buffer (no race: ping-pong) ----
    if (more) {
      const int nxt = cur ^ 1;
      *(uint4*)(&sA[nxt][arow][akc])      = ra0;
      *(uint4*)(&sA[nxt][arow + 64][akc]) = ra1;
      if (TRANSB) {
        *(uint4*)(&sB[nxt][arow][akc])      = rb0;
        *(uint4*)(&sB[nxt][arow + 64][akc]) = rb1;
      } else {
        __bf16 t[8] __attribute__((aligned(16)));
        *(uint4*)t = rb0;
#pragma unroll
        for (int j = 0; j < 8; ++j) sB[nxt][bnc + j][bkr] = t[j];
        *(uint4*)t = rb1;
#pragma unroll
        for (int j = 0; j < 8; ++j) sB[nxt][bnc + j][bkr + 16] = t[j];
      }
    }
  }

  // ---- epilogue ----
  // C/D layout: VGPR v -> row (lane<16 ? v : 8+v), col = lane & 15.
  // A lane owns 8 CONSECUTIVE rows of one column => transposed stores are
  // contiguous (16B bf16 / 32B f32) per fragment.
  const int col_in = lane & 15;
  const int rhalf  = (lane >> 4) << 3;
#pragma unroll
  for (int i = 0; i < 2; ++i) {
#pragma unroll
    for (int j = 0; j < 4; ++j) {
      const long base_row = m0 + wm * 32 + i * 16 + rhalf;
      const long col      = n0 + wn * 64 + j * 16 + col_in;
      if (EPI == 4 || EPI == 5) {
        __bf16 o[8] __attribute__((aligned(16)));
#pragma unroll
        for (int v = 0; v < 8; ++v) {
          float val = acc[i][j][v];
          if (EPI == 5) val = val > 0.0f ? val : 0.0f;
          o[v] = (__bf16)val;
        }
        *(uint4*)((__bf16*)Cout + col * (long)ldc + base_row) = *(const uint4*)o;
      } else if (EPI == 6) {
        float o[8] __attribute__((aligned(16)));
#pragma unroll
        for (int v = 0; v < 8; ++v)
          o[v] = 1.0f / (1.0f + __expf(-acc[i][j][v]));
        float* p = (float*)Cout + col * (long)ldc + base_row;
        *(float4*)(p)     = *(const float4*)(o);
        *(float4*)(p + 4) = *(const float4*)(o + 4);
      } else {
#pragma unroll
        for (int v = 0; v < 8; ++v) {
          float val = acc[i][j][v];
          if (EPI == 0) {
            ((__bf16*)Cout)[(base_row + v) * ldc + col] = (__bf16)val;
          } else {  // EPI == 2
            ((float*)Cout)[(base_row + v) * ldc + col] = val;
          }
        }
      }
    }
  }
}

// ---------------------------------------------------------------------------
// Host-side orchestration
// ---------------------------------------------------------------------------
extern "C" void kernel_launch(void* const* d_in, const int* in_sizes, int n_in,
                              void* d_out, int out_size, void* d_ws, size_t ws_size,
                              hipStream_t stream) {
  (void)in_sizes; (void)n_in; (void)out_size; (void)ws_size;
  const float* x   = (const float*)d_in[0];   // [8192, 512]
  const float* adj = (const float*)d_in[1];   // [8192, 8192]
  const float* W1  = (const float*)d_in[2];   // [512, 256]
  const float* Wm  = (const float*)d_in[3];   // [256, 128]
  const float* Ws  = (const float*)d_in[4];   // [256, 128]
  const float* eps = (const float*)d_in[5];   // [8192, 128]
  float* out = (float*)d_out;                 // [8192, 8192]

  char* ws = (char*)d_ws;
  size_t off = 0;
  auto carve = [&](size_t bytes) -> char* {
    char* p = ws + off;
    off += (bytes + 255) & ~(size_t)255;
    return p;
  };
  __bf16* adjb = (__bf16*)carve((size_t)NN * NN * 2);     // 128 MB
  __bf16* xb   = (__bf16*)carve((size_t)NN * FIN * 2);
  __bf16* w1b  = (__bf16*)carve((size_t)FIN * HH1 * 2);
  __bf16* wmT  = (__bf16*)carve((size_t)HH2 * HH1 * 2);   // Wm^T [128,256]
  __bf16* wsT  = (__bf16*)carve((size_t)HH2 * HH1 * 2);   // Ws^T [128,256]
  __bf16* t0T  = (__bf16*)carve((size_t)HH1 * NN * 2);    // (x@W1)^T [256,8192]
  __bf16* hT   = (__bf16*)carve((size_t)HH1 * NN * 2);    // h^T [256,8192]
  __bf16* hmT  = (__bf16*)carve((size_t)HH2 * NN * 2);    // (h@Wm)^T [128,8192]
  __bf16* hsT  = (__bf16*)carve((size_t)HH2 * NN * 2);    // (h@Ws)^T [128,8192]
  float*  zmf  = (float*)carve((size_t)NN * HH2 * 4);     // adj @ hm  [8192,128]
  float*  zsf  = (float*)carve((size_t)NN * HH2 * 4);     // adj @ hs  [8192,128]
  __bf16* zb   = (__bf16*)carve((size_t)NN * HH2 * 2);    // z [8192,128]

  auto cvt = [&](const float* s, __bf16* d, long n) {
    long n8 = n / 8;
    long blocks = (n8 + 255) / 256;
    if (blocks > 16384) blocks = 16384;
    cvt_f32_bf16_kernel<<<dim3((unsigned)blocks), dim3(256), 0, stream>>>(s, d, n8);
  };

  // 1) precision conversions (adj converted once, reused by both adj-GEMM passes)
  cvt(x,   xb,   (long)NN * FIN);
  cvt(W1,  w1b,  (long)FIN * HH1);
  cvt(adj, adjb, (long)NN * NN);
  cvt_f32_bf16_T_kernel<<<dim3(128), dim3(256), 0, stream>>>(Wm, wmT, HH1, HH2);
  cvt_f32_bf16_T_kernel<<<dim3(128), dim3(256), 0, stream>>>(Ws, wsT, HH1, HH2);

  // 2) t0T = (x @ W1)^T                    [256,8192] bf16
  gemm_bf16_wmma<4, false><<<dim3(NN / BM, HH1 / BN), dim3(256), 0, stream>>>(
      xb, w1b, t0T, FIN, FIN, HH1, NN);

  // 3) hT = relu(adj @ t0)^T               [256,8192] bf16  (B = t0T^T, fast path)
  gemm_bf16_wmma<5, true><<<dim3(NN / BM, HH1 / BN), dim3(256), 0, stream>>>(
      adjb, t0T, hT, NN, NN, NN, NN);

  // 4) hmT = WmT @ hT ; hsT = WsT @ hT     [128,8192] bf16 each
  gemm_bf16_wmma<0, false><<<dim3(HH2 / BM, NN / BN), dim3(256), 0, stream>>>(
      wmT, hT, hmT, HH1, HH1, NN, NN);
  gemm_bf16_wmma<0, false><<<dim3(HH2 / BM, NN / BN), dim3(256), 0, stream>>>(
      wsT, hT, hsT, HH1, HH1, NN, NN);

  // 5) zm = adj @ hm ; zs = adj @ hs       [8192,128] f32  (B = hmT^T, fast path)
  gemm_bf16_wmma<2, true><<<dim3(NN / BM, HH2 / BN), dim3(256), 0, stream>>>(
      adjb, hmT, zmf, NN, NN, NN, HH2);
  gemm_bf16_wmma<2, true><<<dim3(NN / BM, HH2 / BN), dim3(256), 0, stream>>>(
      adjb, hsT, zsf, NN, NN, NN, HH2);

  // 6) z = zm + eps * exp(zs)  -> bf16     [8192,128]
  {
    long n = (long)NN * HH2;
    long blocks = (n + 256L * 4 - 1) / (256L * 4);
    reparam_kernel<<<dim3((unsigned)blocks), dim3(256), 0, stream>>>(zmf, zsf, eps, zb, n);
  }

  // 7) out = sigmoid(z @ z^T)              [8192,8192] f32
  // Result is symmetric, so the transposed (contiguous) store is exact.
  gemm_bf16_wmma<6, true><<<dim3(NN / BM, NN / BN), dim3(256), 0, stream>>>(
      zb, zb, out, HH2, HH2, HH2, NN);
}
